// MultiHeadAttention_56186762166717
// MI455X (gfx1250) — matrix operand order repaired and checked by reading, not compiled, not run on hardware
//
#include <hip/hip_runtime.h>

// ---------------------------------------------------------------- constants
constexpr int kS  = 4096;   // sequence length
constexpr int kD  = 768;    // model dim
constexpr int kH  = 12;     // heads
constexpr int kHD = 64;     // head dim

typedef __bf16 bf16_t;
typedef __attribute__((ext_vector_type(16))) __bf16 v16bf;
typedef __attribute__((ext_vector_type(8)))  __bf16 v8bf;
typedef __attribute__((ext_vector_type(4)))  __bf16 v4bf;
typedef __attribute__((ext_vector_type(8)))  float  v8f;
typedef __attribute__((ext_vector_type(4)))  float  v4f;

// float -> bf16, round to nearest even
__device__ __forceinline__ bf16_t f2bf(float f) {
  unsigned u = __builtin_bit_cast(unsigned, f);
  u += 0x7FFFu + ((u >> 16) & 1u);
  unsigned short h = (unsigned short)(u >> 16);
  return __builtin_bit_cast(bf16_t, h);
}

__device__ __forceinline__ v16bf cat8(v8bf lo, v8bf hi) {
  return __builtin_shufflevector(lo, hi, 0,1,2,3,4,5,6,7,8,9,10,11,12,13,14,15);
}

// Load A-fragment (16x32 bf16): lane m = lane&15,
// e=0..15 -> k = (lane>>4)*16 + e (contiguous, matches B loads)
__device__ __forceinline__ v16bf load_a_frag(const bf16_t* row_base, int hl) {
  v8bf lo = *(const v8bf*)(row_base + hl * 16);
  v8bf hi = *(const v8bf*)(row_base + hl * 16 + 8);
  return cat8(lo, hi);
}

// ---------------------------------------------------------------- kernel 0
// fp32 -> bf16 elementwise conversion, 4 elements per thread
__global__ void cvt_kernel(const float* __restrict__ src, bf16_t* __restrict__ dst,
                           int n4) {
  int i = blockIdx.x * blockDim.x + threadIdx.x;
  int stride = gridDim.x * blockDim.x;
  for (; i < n4; i += stride) {
    v4f f = *(const v4f*)(src + 4 * (size_t)i);
    v4bf b;
#pragma unroll
    for (int j = 0; j < 4; ++j) b[j] = f2bf(f[j]);
    *(v4bf*)(dst + 4 * (size_t)i) = b;
  }
}

// ---------------------------------------------------------------- kernel 1
// Q/K/V projection: y[s,o] = sum_d x[s,d] * W[o,d]   (Linear: x @ W^T)
// One wave computes a 16x64 output tile of one of {Q,K,V} (A reused 4x).
// Q gets the softmax scale (0.125) folded in.  V is stored transposed: Vt[o][s].
__global__ __launch_bounds__(128)
void qkv_kernel(const bf16_t* __restrict__ xb,
                const bf16_t* __restrict__ wq,
                const bf16_t* __restrict__ wk,
                const bf16_t* __restrict__ wv,
                bf16_t* __restrict__ qout,
                bf16_t* __restrict__ kout,
                bf16_t* __restrict__ vt) {
  const int lane = threadIdx.x & 31;
  const int wave = threadIdx.x >> 5;
  int tile = blockIdx.x * 4 + wave;              // 3 * (S/16) * (D/64) tiles
  const int tiles_per_mat = (kS / 16) * (kD / 64);   // 3072
  const int mat = tile / tiles_per_mat;
  tile %= tiles_per_mat;
  const int s0 = (tile / (kD / 64)) * 16;
  const int o0 = (tile % (kD / 64)) * 64;
  const int hl = lane >> 4, l16 = lane & 15;

  const bf16_t* w = (mat == 0) ? wq : (mat == 1) ? wk : wv;

  v8f c[4] = {};
  const bf16_t* arow = xb + (size_t)(s0 + l16) * kD;
  const bf16_t* brow = w  + (size_t)(o0 + l16) * kD + hl * 16;
  for (int k0 = 0; k0 < kD; k0 += 32) {
    v16bf a = load_a_frag(arow + k0, hl);
#pragma unroll
    for (int j = 0; j < 4; ++j) {
      v16bf b = *(const v16bf*)(brow + (size_t)j * 16 * kD + k0);
      c[j] = __builtin_amdgcn_wmma_f32_16x16x32_bf16(false, a, false, b, (short)0,
                                                     c[j], false, false);
    }
  }

#pragma unroll
  for (int j = 0; j < 4; ++j)
#pragma unroll
    for (int r = 0; r < 8; ++r) {
      const int m = r + 8 * hl;                  // C frag: row = r + 8*half
      const int o = o0 + j * 16 + l16;
      const float v = c[j][r];
      if (mat == 0)
        qout[(size_t)(s0 + m) * kD + o] = f2bf(v * 0.125f);   // 1/sqrt(64)
      else if (mat == 1)
        kout[(size_t)(s0 + m) * kD + o] = f2bf(v);
      else
        vt[(size_t)o * kS + s0 + m] = f2bf(v);                // transposed
    }
}

// ---------------------------------------------------------------- kernel 2
// Flash attention, causal.  One wave per (head, 32-query tile), 32-key blocks.
// Two 16-row query sub-tiles share each K/V fragment load (halves K/V traffic).
__global__ __launch_bounds__(128)
void fa_kernel(const bf16_t* __restrict__ qb,
               const bf16_t* __restrict__ kb,
               const bf16_t* __restrict__ vt,
               bf16_t* __restrict__ ctx) {
  __shared__ __align__(16) bf16_t pbuf[4][32 * 32];  // per-wave P staging

  const int lane = threadIdx.x & 31;
  const int wave = threadIdx.x >> 5;
  const int tile = blockIdx.x * 4 + wave;        // kH * (kS/32) tiles
  const int h  = tile / (kS / 32);
  const int q0 = (tile % (kS / 32)) * 32;
  const int hl = lane >> 4, l16 = lane & 15;

  // Q A-fragments: 2 query sub-tiles x 2 d-chunks
  v16bf aq[2][2];
#pragma unroll
  for (int qs = 0; qs < 2; ++qs)
#pragma unroll
    for (int dc = 0; dc < 2; ++dc)
      aq[qs][dc] = load_a_frag(
          qb + (size_t)(q0 + qs * 16 + l16) * kD + h * kHD + dc * 32, hl);

  float mrow[2][8], lrow[2][8];
  v8f acc[2][4] = {};
#pragma unroll
  for (int qs = 0; qs < 2; ++qs)
#pragma unroll
    for (int r = 0; r < 8; ++r) { mrow[qs][r] = -3.0e38f; lrow[qs][r] = 0.0f; }

  for (int kb0 = 0; kb0 < q0 + 32; kb0 += 32) {   // causal: keys <= q0+31
    // ---- load K B-fragments once, shared by both query sub-tiles
    v16bf bk[2][2];
#pragma unroll
    for (int nc = 0; nc < 2; ++nc)
#pragma unroll
      for (int dc = 0; dc < 2; ++dc)
        bk[nc][dc] = *(const v16bf*)(kb + (size_t)(kb0 + nc * 16 + l16) * kD
                                        + h * kHD + dc * 32 + hl * 16);
    // prefetch next key block (K rows + Vt rows) while we do math
    if (kb0 + 32 < q0 + 32) {
#pragma unroll
      for (int nc = 0; nc < 2; ++nc)
        __builtin_prefetch(kb + (size_t)(kb0 + 32 + nc * 16 + l16) * kD + h * kHD,
                           0, 3);
      __builtin_prefetch(vt + (size_t)(h * kHD + l16) * kS + kb0 + 32, 0, 3);
    }
    // ---- S = Q * K^T  (two 16x32 score tiles)
    v8f s[2][2] = {};
#pragma unroll
    for (int qs = 0; qs < 2; ++qs)
#pragma unroll
      for (int nc = 0; nc < 2; ++nc)
#pragma unroll
        for (int dc = 0; dc < 2; ++dc)
          s[qs][nc] = __builtin_amdgcn_wmma_f32_16x16x32_bf16(
              false, aq[qs][dc], false, bk[nc][dc], (short)0, s[qs][nc],
              false, false);
    // ---- causal mask + streaming softmax per sub-tile
#pragma unroll
    for (int qs = 0; qs < 2; ++qs) {
#pragma unroll
      for (int nc = 0; nc < 2; ++nc)
#pragma unroll
        for (int r = 0; r < 8; ++r) {
          const int col = kb0 + nc * 16 + l16;
          const int row = q0 + qs * 16 + r + 8 * hl;
          if (col > row) s[qs][nc][r] = -3.0e38f;
        }
#pragma unroll
      for (int r = 0; r < 8; ++r) {
        float bm = fmaxf(s[qs][0][r], s[qs][1][r]);
#pragma unroll
        for (int off = 8; off >= 1; off >>= 1)
          bm = fmaxf(bm, __shfl_xor(bm, off, 16));
        const float mn    = fmaxf(mrow[qs][r], bm);
        const float alpha = __expf(mrow[qs][r] - mn);
        const float p0 = __expf(s[qs][0][r] - mn);
        const float p1 = __expf(s[qs][1][r] - mn);
        s[qs][0][r] = p0; s[qs][1][r] = p1;
        float rs = p0 + p1;
#pragma unroll
        for (int off = 8; off >= 1; off >>= 1) rs += __shfl_xor(rs, off, 16);
        lrow[qs][r] = lrow[qs][r] * alpha + rs;
        mrow[qs][r] = mn;
#pragma unroll
        for (int nd = 0; nd < 4; ++nd) acc[qs][nd][r] *= alpha;
      }
      // write P sub-tile to LDS (C-layout -> memory row-major)
#pragma unroll
      for (int nc = 0; nc < 2; ++nc)
#pragma unroll
        for (int r = 0; r < 8; ++r)
          pbuf[wave][(qs * 16 + r + 8 * hl) * 32 + nc * 16 + l16] =
              f2bf(s[qs][nc][r]);
    }
    asm volatile("s_wait_dscnt 0x0" ::: "memory");   // per-wave LDS RAW
    // ---- load V B-fragments once, shared by both query sub-tiles
    v16bf bv[4];
#pragma unroll
    for (int nd = 0; nd < 4; ++nd)
      bv[nd] = *(const v16bf*)(vt + (size_t)(h * kHD + nd * 16 + l16) * kS
                                  + kb0 + hl * 16);
    // ---- acc += P * V
#pragma unroll
    for (int qs = 0; qs < 2; ++qs) {
      v16bf pa = load_a_frag(&pbuf[wave][(qs * 16 + l16) * 32], hl);
#pragma unroll
      for (int nd = 0; nd < 4; ++nd)
        acc[qs][nd] = __builtin_amdgcn_wmma_f32_16x16x32_bf16(
            false, pa, false, bv[nd], (short)0, acc[qs][nd], false, false);
    }
  }

  // ---- epilogue: normalize and store ctx (bf16)
#pragma unroll
  for (int qs = 0; qs < 2; ++qs)
#pragma unroll
    for (int r = 0; r < 8; ++r) {
      const float inv = 1.0f / lrow[qs][r];
      const int m = q0 + qs * 16 + r + 8 * hl;
#pragma unroll
      for (int nd = 0; nd < 4; ++nd)
        ctx[(size_t)m * kD + h * kHD + nd * 16 + l16] = f2bf(acc[qs][nd][r] * inv);
    }
}

// ---------------------------------------------------------------- kernel 3
// Output projection: out[s,o] = sum_d ctx[s,d] * Wo[o,d] + b_o[o], fp32 out
// One wave computes a 16x64 output tile (A reused 4x).
__global__ __launch_bounds__(128)
void oproj_kernel(const bf16_t* __restrict__ ctx,
                  const bf16_t* __restrict__ wo,
                  const float* __restrict__ bo,
                  float* __restrict__ out) {
  const int lane = threadIdx.x & 31;
  const int wave = threadIdx.x >> 5;
  const int tile = blockIdx.x * 4 + wave;        // (S/16)*(D/64) tiles
  const int s0 = (tile / (kD / 64)) * 16;
  const int o0 = (tile % (kD / 64)) * 64;
  const int hl = lane >> 4, l16 = lane & 15;

  v8f c[4] = {};
  const bf16_t* arow = ctx + (size_t)(s0 + l16) * kD;
  const bf16_t* brow = wo  + (size_t)(o0 + l16) * kD + hl * 16;
  for (int k0 = 0; k0 < kD; k0 += 32) {
    v16bf a = load_a_frag(arow + k0, hl);
#pragma unroll
    for (int j = 0; j < 4; ++j) {
      v16bf b = *(const v16bf*)(brow + (size_t)j * 16 * kD + k0);
      c[j] = __builtin_amdgcn_wmma_f32_16x16x32_bf16(false, a, false, b, (short)0,
                                                     c[j], false, false);
    }
  }
#pragma unroll
  for (int j = 0; j < 4; ++j)
#pragma unroll
    for (int r = 0; r < 8; ++r) {
      const int m = r + 8 * hl;
      const int o = o0 + j * 16 + l16;
      out[(size_t)(s0 + m) * kD + o] = c[j][r] + bo[o];
    }
}

// ---------------------------------------------------------------- launch
extern "C" void kernel_launch(void* const* d_in, const int* in_sizes, int n_in,
                              void* d_out, int out_size, void* d_ws, size_t ws_size,
                              hipStream_t stream) {
  (void)in_sizes; (void)n_in; (void)out_size; (void)ws_size;
  // setup_inputs order: x, w_k, w_q, w_v, w_o, b_o
  const float* x  = (const float*)d_in[0];
  const float* wk = (const float*)d_in[1];
  const float* wq = (const float*)d_in[2];
  const float* wv = (const float*)d_in[3];
  const float* wo = (const float*)d_in[4];
  const float* bo = (const float*)d_in[5];
  float* out = (float*)d_out;

  const size_t SD = (size_t)kS * kD;   // 3,145,728
  const size_t DD = (size_t)kD * kD;   //   589,824
  bf16_t* ws  = (bf16_t*)d_ws;         // ~36.2 MB of bf16 scratch
  bf16_t* xb  = ws;
  bf16_t* wqb = xb  + SD;
  bf16_t* wkb = wqb + DD;
  bf16_t* wvb = wkb + DD;
  bf16_t* wob = wvb + DD;
  bf16_t* qb  = wob + DD;
  bf16_t* kbf = qb  + SD;
  bf16_t* vtb = kbf + SD;   // V transposed: [D][S]
  bf16_t* ctx = vtb + SD;

  // 0) fp32 -> bf16 conversions (4 elems/thread)
  cvt_kernel<<<512, 256, 0, stream>>>(x,  xb,  (int)(SD / 4));
  cvt_kernel<<<256, 256, 0, stream>>>(wq, wqb, (int)(DD / 4));
  cvt_kernel<<<256, 256, 0, stream>>>(wk, wkb, (int)(DD / 4));
  cvt_kernel<<<256, 256, 0, stream>>>(wv, wvb, (int)(DD / 4));
  cvt_kernel<<<256, 256, 0, stream>>>(wo, wob, (int)(DD / 4));

  // 1) QKV projections: 3 * 256 * 12 = 9216 waves, 4 waves/block
  qkv_kernel<<<2304, 128, 0, stream>>>(xb, wqb, wkb, wvb, qb, kbf, vtb);

  // 2) flash attention: 12 * 128 = 1536 waves, 4 waves/block
  fa_kernel<<<384, 128, 0, stream>>>(qb, kbf, vtb, ctx);

  // 3) output projection: 256 * 12 = 3072 waves, 4 waves/block
  oproj_kernel<<<768, 128, 0, stream>>>(ctx, wob, bo, out);
}
